// CaptionEmbedding_46986942218474
// MI455X (gfx1250) — matrix operand order repaired and checked by reading, not compiled, not run on hardware
//
#include <hip/hip_runtime.h>

// Problem constants (fixed by the reference)
#define Bsz 512
#define VD  2048
#define QD  1024
#define HD  1024
#define LL  20
#define G3  3072   // 3*HD

typedef unsigned short bf16;  // raw bf16 bits
typedef __attribute__((ext_vector_type(16))) __bf16 bf16x16;
typedef __attribute__((ext_vector_type(8)))  float  f32x8;

__device__ __forceinline__ bf16 f2bf(float f) {
  unsigned u = __float_as_uint(f);
  unsigned r = 0x7FFFu + ((u >> 16) & 1u);   // round-to-nearest-even
  return (bf16)((u + r) >> 16);
}
__device__ __forceinline__ float sigmf(float x) { return 1.f / (1.f + expf(-x)); }

// CDNA5 async DMA: per-lane 16B global -> LDS, tracked by ASYNCcnt.
__device__ __forceinline__ void async_ld_b128(unsigned lds_off, const bf16* g) {
  asm volatile("global_load_async_to_lds_b128 %0, %1, off"
               :: "v"(lds_off), "v"((unsigned long long)(const void*)g)
               : "memory");
}
__device__ __forceinline__ void wait_async0() {
  asm volatile("s_wait_asynccnt 0x0" ::: "memory");
}
__device__ __forceinline__ unsigned lds_off_of(const void* p) {
  // generic shared-aperture address keeps the LDS byte offset in addr[31:0]
  return (unsigned)(unsigned long long)p;
}

// ---------------------------------------------------------------------------
// Tiled bf16 WMMA GEMM:  out = epilogue( A[M,K] @ Bw[N,K]^T + bias )
// Workgroup: 256 threads = 8 waves, tile 128(M) x 64(N), K chunks of 32,
// double-buffered LDS staged by global_load_async_to_lds_b128 (ASYNCcnt).
// Wave grid 4x2; each wave owns a 32x32 block = 2x2 tiles of 16x16.
// EPI 0: outF = acc + bias (+ addend)                      (f32 out)
// EPI 1: outBf = bf16( relu(acc + bias + addend) )         (joint)
// EPI 2: a = sigmoid(acc+bias); alphas[.,t,.] = m? a : 0;
//        attxBf = bf16(a * caption[order[row],t,col])      (att)
// EPI 3: h2n = acc+bias; masked h2 state update + bf16 copy;
//        outs[.,t,.] = m? h2n : 0                          (fc)
// ---------------------------------------------------------------------------
template <int EPI>
__global__ __launch_bounds__(256)
void gemm_bf16_wmma(const bf16* __restrict__ A, int lda,
                    const bf16* __restrict__ Bw, int K, int N,
                    const float* __restrict__ bias,
                    const float* __restrict__ addend,
                    float* outF,
                    bf16*  outBf,
                    float* alphasOut,
                    bf16*  attxBf,
                    const float* __restrict__ capF,
                    const int* __restrict__ order,
                    const int* __restrict__ cls,
                    float* h2state,
                    bf16*  h2bf,
                    float* outsP,
                    int t) {
  __shared__ __align__(16) bf16 As[2][128 * 32];   // 2 x 8 KB
  __shared__ __align__(16) bf16 Bs[2][64 * 32];    // 2 x 4 KB

  const int tid  = threadIdx.x;
  const int wave = tid >> 5;
  const int lane = tid & 31;
  const int wm = wave >> 1;        // 0..3 : wave M position
  const int wn = wave & 1;         // 0..1 : wave N position
  const int rowA0 = blockIdx.x * 128;
  const int colB0 = blockIdx.y * 64;
  const int lr = lane & 15;
  const int lh = lane >> 4;

  f32x8 acc[2][2] = {};

  // global->LDS staging assignments (per thread: A 32B = 2 x b128, B 16B = 1 x b128)
  const int ar = tid >> 1;          // 0..127 A row in tile
  const int ac = (tid & 1) * 16;    // 0 | 16
  const int br = tid >> 2;          // 0..63  B row in tile
  const int bc = (tid & 3) * 8;     // 0/8/16/24

  const bf16* aSrc = A  + (size_t)(rowA0 + ar) * lda + ac;
  const bf16* bSrc = Bw + (size_t)(colB0 + br) * K   + bc;
  const unsigned aDst[2] = { lds_off_of(&As[0][ar * 32 + ac]),
                             lds_off_of(&As[1][ar * 32 + ac]) };
  const unsigned bDst[2] = { lds_off_of(&Bs[0][br * 32 + bc]),
                             lds_off_of(&Bs[1][br * 32 + bc]) };

  const int nk = K >> 5;
  // prologue: stage chunk 0 into buffer 0
  async_ld_b128(aDst[0],      aSrc);
  async_ld_b128(aDst[0] + 16, aSrc + 8);
  async_ld_b128(bDst[0],      bSrc);

  for (int i = 0; i < nk; ++i) {
    wait_async0();     // my chunk-i transfers landed
    __syncthreads();   // everyone's landed; all chunk-(i-1) LDS reads retired
    if (i + 1 < nk) {  // overlap next chunk's DMA with this chunk's WMMAs
      const int nb = (i + 1) & 1;
      const int k1 = (i + 1) << 5;
      async_ld_b128(aDst[nb],      aSrc + k1);
      async_ld_b128(aDst[nb] + 16, aSrc + k1 + 8);
      async_ld_b128(bDst[nb],      bSrc + k1);
    }
    const int buf = i & 1;

    // fragments per ISA VGPR layout (16-bit A 16x32 / B 32x16)
    union { uint4 u[2]; bf16x16 v; } fa[2], fb[2];
#pragma unroll
    for (int mt = 0; mt < 2; ++mt) {
      const uint4* p = reinterpret_cast<const uint4*>(&As[buf][(wm * 32 + mt * 16 + lr) * 32 + lh * 8]);
      fa[mt].u[0] = p[0];   // K = lh*8 .. +7
      fa[mt].u[1] = p[2];   // K = 16 + lh*8 .. +7
    }
#pragma unroll
    for (int nt = 0; nt < 2; ++nt) {
      const uint4* p = reinterpret_cast<const uint4*>(&Bs[buf][(wn * 32 + nt * 16 + lr) * 32 + lh * 8]);
      fb[nt].u[0] = p[0];   // K = lh*8 .. +7
      fb[nt].u[1] = p[2];   // K = 16 + lh*8 .. +7
    }
#pragma unroll
    for (int mt = 0; mt < 2; ++mt)
#pragma unroll
      for (int nt = 0; nt < 2; ++nt)
        acc[mt][nt] = __builtin_amdgcn_wmma_f32_16x16x32_bf16(
            false, fa[mt].v, false, fb[nt].v, (short)0, acc[mt][nt], false, false);
  }

  // epilogue: VGPR r -> row = 8*lh + r ; col = lr
#pragma unroll
  for (int mt = 0; mt < 2; ++mt) {
#pragma unroll
    for (int nt = 0; nt < 2; ++nt) {
      const int col = colB0 + wn * 32 + nt * 16 + lr;
      const float bv = bias ? bias[col] : 0.f;
#pragma unroll
      for (int r = 0; r < 8; ++r) {
        const int row = rowA0 + wm * 32 + mt * 16 + lh * 8 + r;
        float x = acc[mt][nt][r] + bv;
        const size_t oi = (size_t)row * N + col;
        if (EPI == 0) {
          if (addend) x += addend[oi];
          outF[oi] = x;
        } else if (EPI == 1) {
          x += addend[oi];                       // + pvq (original batch order!)
          outBf[oi] = f2bf(fmaxf(x, 0.f));
        } else if (EPI == 2) {
          const float a = sigmf(x);
          const bool  m = t < cls[row];
          alphasOut[(size_t)row * (LL * HD) + (size_t)t * HD + col] = m ? a : 0.f;
          const float xf = capF[(size_t)order[row] * (LL * QD) + (size_t)t * QD + col];
          attxBf[(size_t)row * HD + col] = f2bf(a * xf);
        } else {  // EPI == 3
          const bool m = t < cls[row];
          const size_t hi = (size_t)row * HD + col;
          const float hold = h2state[hi];
          const float hnew = m ? x : hold;
          h2state[hi] = hnew;
          h2bf[hi]    = f2bf(hnew);
          outsP[(size_t)row * (LL * HD) + (size_t)t * HD + col] = m ? x : 0.f;
        }
      }
    }
  }
}

// ---------------------------------------------------------------------------
// Elementwise GRU gate kernels (fp32 state math)
// ---------------------------------------------------------------------------
__global__ void gru_step1(const float* __restrict__ gi, const float* __restrict__ gh,
                          float* __restrict__ h, bf16* __restrict__ hbf,
                          const int* __restrict__ cls, int t) {
  const int idx = blockIdx.x * 256 + threadIdx.x;    // [0, Bsz*HD)
  const int row = idx >> 10;
  const int j   = idx & 1023;
  const size_t b3 = (size_t)row * G3;
  const float r = sigmf(gi[b3 + j] + gh[b3 + j]);
  const float z = sigmf(gi[b3 + HD + j] + gh[b3 + HD + j]);
  const float n = tanhf(gi[b3 + 2 * HD + j] + r * gh[b3 + 2 * HD + j]);
  const float hc = h[idx];
  const float o  = (1.f - z) * n + z * hc;
  const float hn = (t < cls[row]) ? o : hc;
  h[idx]   = hn;
  hbf[idx] = f2bf(hn);
}

__global__ void gru_step2(const float* __restrict__ gi, const float* __restrict__ gh,
                          const float* __restrict__ h2, bf16* __restrict__ gbf) {
  const int idx = blockIdx.x * 256 + threadIdx.x;
  const int row = idx >> 10;
  const int j   = idx & 1023;
  const size_t b3 = (size_t)row * G3;
  const float r = sigmf(gi[b3 + j] + gh[b3 + j]);
  const float z = sigmf(gi[b3 + HD + j] + gh[b3 + HD + j]);
  const float n = tanhf(gi[b3 + 2 * HD + j] + r * gh[b3 + 2 * HD + j]);
  const float o = (1.f - z) * n + z * h2[idx];   // unmasked: mask applied at fc
  gbf[idx] = f2bf(o);
}

// ---------------------------------------------------------------------------
// Setup kernels
// ---------------------------------------------------------------------------
__global__ void zero_u32(unsigned* p, int n) {
  int i = blockIdx.x * 256 + threadIdx.x;
  if (i < n) p[i] = 0u;
}

__global__ void sumsq(const float* __restrict__ x, int n, float* out) {
  __shared__ float red[256];
  float s = 0.f;
  for (int i = blockIdx.x * blockDim.x + threadIdx.x; i < n; i += gridDim.x * blockDim.x) {
    float v = x[i]; s += v * v;
  }
  red[threadIdx.x] = s;
  __syncthreads();
  for (int o = 128; o > 0; o >>= 1) {
    if ((int)threadIdx.x < o) red[threadIdx.x] += red[threadIdx.x + o];
    __syncthreads();
  }
  if (threadIdx.x == 0) atomicAdd(out, red[0]);
}

__global__ void cvt_scaled(const float* __restrict__ V, const float* __restrict__ g,
                           const float* __restrict__ sum, bf16* __restrict__ o, int n) {
  int i = blockIdx.x * 256 + threadIdx.x;
  if (i >= n) return;
  const float sc = g[0] * rsqrtf(sum[0]);   // g / ||V||_F
  o[i] = f2bf(V[i] * sc);
}

__global__ void cvt_plain(const float* __restrict__ x, bf16* __restrict__ o, int n) {
  int i = blockIdx.x * 256 + threadIdx.x;
  if (i < n) o[i] = f2bf(x[i]);
}

// stable counting sort by descending length (lengths in [1, LL])
__global__ void sort_len(const int* __restrict__ len, int* __restrict__ order,
                         int* __restrict__ cls) {
  if (threadIdx.x != 0 || blockIdx.x != 0) return;
  int pos = 0;
  for (int v = LL; v >= 1; --v)
    for (int i = 0; i < Bsz; ++i)
      if (len[i] == v) { order[pos] = i; cls[pos] = v; ++pos; }
}

__global__ void cap_sort_cvt(const float* __restrict__ cap, const int* __restrict__ order,
                             bf16* __restrict__ o) {
  int i = blockIdx.x * 256 + threadIdx.x;
  if (i >= Bsz * LL * QD) return;
  const int s = i / (LL * QD);
  const int r = i - s * (LL * QD);
  o[i] = f2bf(cap[(size_t)order[s] * (LL * QD) + r]);
}

// ---------------------------------------------------------------------------
extern "C" void kernel_launch(void* const* d_in, const int* in_sizes, int n_in,
                              void* d_out, int out_size, void* d_ws, size_t ws_size,
                              hipStream_t stream) {
  (void)in_sizes; (void)n_in; (void)out_size; (void)ws_size;
  const float* vI  = (const float*)d_in[0];
  const float* qI  = (const float*)d_in[1];
  const float* capI = (const float*)d_in[2];
  const int*   capLen = (const int*)d_in[3];
  const float* wWih = (const float*)d_in[4];
  const float* wWhh = (const float*)d_in[5];
  const float* wBih = (const float*)d_in[6];
  const float* wBhh = (const float*)d_in[7];
  const float* cWih = (const float*)d_in[8];
  const float* cWhh = (const float*)d_in[9];
  const float* cBih = (const float*)d_in[10];
  const float* cBhh = (const float*)d_in[11];
  const float* avV = (const float*)d_in[12]; const float* avG = (const float*)d_in[13]; const float* avB = (const float*)d_in[14];
  const float* aqV = (const float*)d_in[15]; const float* aqG = (const float*)d_in[16]; const float* aqB = (const float*)d_in[17];
  const float* ahV = (const float*)d_in[18]; const float* ahG = (const float*)d_in[19]; const float* ahB = (const float*)d_in[20];
  const float* alV = (const float*)d_in[21]; const float* alG = (const float*)d_in[22]; const float* alB = (const float*)d_in[23];
  const float* fcV = (const float*)d_in[24]; const float* fcG = (const float*)d_in[25]; const float* fcB = (const float*)d_in[26];

  float* outs   = (float*)d_out;
  float* alphas = outs + (size_t)Bsz * LL * HD;

  // workspace carve-out (~96 MiB), 256B aligned blocks
  char* base = (char*)d_ws;
  size_t off = 0;
  auto alloc = [&](size_t bytes) -> void* {
    void* p = base + off;
    off = (off + bytes + 255) & ~((size_t)255);
    return p;
  };
  float* sums  = (float*)alloc(8 * sizeof(float));
  int*   order = (int*)alloc(Bsz * sizeof(int));
  int*   cls   = (int*)alloc(Bsz * sizeof(int));
  bf16* Wv   = (bf16*)alloc((size_t)HD * VD * 2);
  bf16* Wq   = (bf16*)alloc((size_t)HD * QD * 2);
  bf16* Wh   = (bf16*)alloc((size_t)HD * HD * 2);
  bf16* Wl   = (bf16*)alloc((size_t)HD * HD * 2);
  bf16* Wf   = (bf16*)alloc((size_t)HD * HD * 2);
  bf16* Wih1 = (bf16*)alloc((size_t)G3 * QD * 2);
  bf16* Whh1 = (bf16*)alloc((size_t)G3 * HD * 2);
  bf16* Wih2 = (bf16*)alloc((size_t)G3 * HD * 2);
  bf16* Whh2 = (bf16*)alloc((size_t)G3 * HD * 2);
  bf16* vBf  = (bf16*)alloc((size_t)Bsz * VD * 2);
  bf16* qBf  = (bf16*)alloc((size_t)Bsz * QD * 2);
  bf16* capBf = (bf16*)alloc((size_t)Bsz * LL * QD * 2);
  float* pvqA = (float*)alloc((size_t)Bsz * HD * 4);
  float* pvq  = (float*)alloc((size_t)Bsz * HD * 4);
  float* h1   = (float*)alloc((size_t)Bsz * HD * 4);
  float* h2   = (float*)alloc((size_t)Bsz * HD * 4);
  bf16* h1Bf    = (bf16*)alloc((size_t)Bsz * HD * 2);
  bf16* h2Bf    = (bf16*)alloc((size_t)Bsz * HD * 2);
  bf16* jointBf = (bf16*)alloc((size_t)Bsz * HD * 2);
  bf16* attxBf  = (bf16*)alloc((size_t)Bsz * HD * 2);
  bf16* g2Bf    = (bf16*)alloc((size_t)Bsz * HD * 2);
  float* gi1 = (float*)alloc((size_t)Bsz * G3 * 4);
  float* gh1 = (float*)alloc((size_t)Bsz * G3 * 4);
  float* gi2 = (float*)alloc((size_t)Bsz * G3 * 4);
  float* gh2 = (float*)alloc((size_t)Bsz * G3 * 4);

  // ---- setup: norms, sort, bf16 staging (all deterministic, re-done per call)
  zero_u32<<<1, 256, 0, stream>>>((unsigned*)sums, 8);
  sumsq<<<256, 256, 0, stream>>>(avV, HD * VD, sums + 0);
  sumsq<<<256, 256, 0, stream>>>(aqV, HD * QD, sums + 1);
  sumsq<<<256, 256, 0, stream>>>(ahV, HD * HD, sums + 2);
  sumsq<<<256, 256, 0, stream>>>(alV, HD * HD, sums + 3);
  sumsq<<<256, 256, 0, stream>>>(fcV, HD * HD, sums + 4);
  sort_len<<<1, 1, 0, stream>>>(capLen, order, cls);
  cvt_scaled<<<(HD * VD) / 256, 256, 0, stream>>>(avV, avG, sums + 0, Wv, HD * VD);
  cvt_scaled<<<(HD * QD) / 256, 256, 0, stream>>>(aqV, aqG, sums + 1, Wq, HD * QD);
  cvt_scaled<<<(HD * HD) / 256, 256, 0, stream>>>(ahV, ahG, sums + 2, Wh, HD * HD);
  cvt_scaled<<<(HD * HD) / 256, 256, 0, stream>>>(alV, alG, sums + 3, Wl, HD * HD);
  cvt_scaled<<<(HD * HD) / 256, 256, 0, stream>>>(fcV, fcG, sums + 4, Wf, HD * HD);
  cvt_plain<<<(G3 * QD) / 256, 256, 0, stream>>>(wWih, Wih1, G3 * QD);
  cvt_plain<<<(G3 * HD) / 256, 256, 0, stream>>>(wWhh, Whh1, G3 * HD);
  cvt_plain<<<(G3 * HD) / 256, 256, 0, stream>>>(cWih, Wih2, G3 * HD);
  cvt_plain<<<(G3 * HD) / 256, 256, 0, stream>>>(cWhh, Whh2, G3 * HD);
  cvt_plain<<<(Bsz * VD) / 256, 256, 0, stream>>>(vI, vBf, Bsz * VD);
  cvt_plain<<<(Bsz * QD) / 256, 256, 0, stream>>>(qI, qBf, Bsz * QD);
  cap_sort_cvt<<<(Bsz * LL * QD) / 256, 256, 0, stream>>>(capI, order, capBf);
  zero_u32<<<(Bsz * HD) / 256, 256, 0, stream>>>((unsigned*)h1, Bsz * HD);
  zero_u32<<<(Bsz * HD) / 256, 256, 0, stream>>>((unsigned*)h2, Bsz * HD);
  zero_u32<<<(Bsz * HD / 2) / 256, 256, 0, stream>>>((unsigned*)h1Bf, Bsz * HD / 2);
  zero_u32<<<(Bsz * HD / 2) / 256, 256, 0, stream>>>((unsigned*)h2Bf, Bsz * HD / 2);

#define GRID(N_) dim3(Bsz / 128, (N_) / 64)
  // pvq = v@Wv.T + av_b + q@Wq.T + aq_b   (ORIGINAL batch order, per reference)
  gemm_bf16_wmma<0><<<GRID(HD), 256, 0, stream>>>(vBf, VD, Wv, VD, HD, avB, nullptr, pvqA,
      nullptr, nullptr, nullptr, nullptr, nullptr, nullptr, nullptr, nullptr, nullptr, 0);
  gemm_bf16_wmma<0><<<GRID(HD), 256, 0, stream>>>(qBf, QD, Wq, QD, HD, aqB, pvqA, pvq,
      nullptr, nullptr, nullptr, nullptr, nullptr, nullptr, nullptr, nullptr, nullptr, 0);

  for (int t = 0; t < LL; ++t) {
    // GRU1 gates
    gemm_bf16_wmma<0><<<GRID(G3), 256, 0, stream>>>(capBf + (size_t)t * QD, LL * QD, Wih1, QD, G3,
        wBih, nullptr, gi1, nullptr, nullptr, nullptr, nullptr, nullptr, nullptr, nullptr, nullptr, nullptr, t);
    gemm_bf16_wmma<0><<<GRID(G3), 256, 0, stream>>>(h1Bf, HD, Whh1, HD, G3,
        wBhh, nullptr, gh1, nullptr, nullptr, nullptr, nullptr, nullptr, nullptr, nullptr, nullptr, nullptr, t);
    gru_step1<<<(Bsz * HD) / 256, 256, 0, stream>>>(gi1, gh1, h1, h1Bf, cls, t);
    // joint = relu(pvq + h1@Wh.T + ah_b)
    gemm_bf16_wmma<1><<<GRID(HD), 256, 0, stream>>>(h1Bf, HD, Wh, HD, HD,
        ahB, pvq, nullptr, jointBf, nullptr, nullptr, nullptr, nullptr, nullptr, nullptr, nullptr, nullptr, t);
    // att = sigmoid(joint@Wl.T + al_b); alphas + att*x_t
    gemm_bf16_wmma<2><<<GRID(HD), 256, 0, stream>>>(jointBf, HD, Wl, HD, HD,
        alB, nullptr, nullptr, nullptr, alphas, attxBf, capI, order, cls, nullptr, nullptr, nullptr, t);
    // GRU2 gates
    gemm_bf16_wmma<0><<<GRID(G3), 256, 0, stream>>>(attxBf, HD, Wih2, HD, G3,
        cBih, nullptr, gi2, nullptr, nullptr, nullptr, nullptr, nullptr, nullptr, nullptr, nullptr, nullptr, t);
    gemm_bf16_wmma<0><<<GRID(G3), 256, 0, stream>>>(h2Bf, HD, Whh2, HD, G3,
        cBhh, nullptr, gh2, nullptr, nullptr, nullptr, nullptr, nullptr, nullptr, nullptr, nullptr, nullptr, t);
    gru_step2<<<(Bsz * HD) / 256, 256, 0, stream>>>(gi2, gh2, h2, g2Bf);
    // h2n = g2@Wf.T + fc_b ; masked state update + outs store
    gemm_bf16_wmma<3><<<GRID(HD), 256, 0, stream>>>(g2Bf, HD, Wf, HD, HD,
        fcB, nullptr, nullptr, nullptr, nullptr, nullptr, nullptr, nullptr, cls, h2, h2Bf, outs, t);
  }
#undef GRID
}
